// VisualAttention_56813827392103
// MI455X (gfx1250) — matrix operand order repaired and checked by reading, not compiled, not run on hardware
//
#include <hip/hip_runtime.h>
#include <hip/hip_bf16.h>

typedef unsigned short u16;
typedef unsigned int   u32;

typedef __attribute__((ext_vector_type(16))) __bf16 v16bf;
typedef __attribute__((ext_vector_type(8)))  __bf16 v8bf;
typedef __attribute__((ext_vector_type(8)))  float  v8f;
typedef int vec4i __attribute__((vector_size(4 * sizeof(int))));   // b128 payload type

#define AS1 __attribute__((address_space(1)))
#define AS3 __attribute__((address_space(3)))

// CDNA5 async global->LDS path (guarded; falls back to load+ds_store)
#if defined(__gfx1250__) && __has_builtin(__builtin_amdgcn_global_load_async_to_lds_b128)
#define HAVE_ASYNC_LDS 1
#else
#define HAVE_ASYNC_LDS 0
#endif

#if HAVE_ASYNC_LDS
#if __has_builtin(__builtin_amdgcn_s_wait_asynccnt)
#define WAIT_ASYNC() __builtin_amdgcn_s_wait_asynccnt(0)
#else
#define WAIT_ASYNC() asm volatile("s_wait_asynccnt 0x0" ::: "memory")
#endif
#endif

// ---------- helpers ----------
__device__ __forceinline__ u16 f2bf(float f) {
    u32 u = __float_as_uint(f);
    u = (u + 0x7FFFu + ((u >> 16) & 1u)) >> 16;   // round-to-nearest-even
    return (u16)u;
}
__device__ __forceinline__ float bf2f(u16 h) {
    return __uint_as_float(((u32)h) << 16);
}
__device__ __forceinline__ v16bf cat8(v8bf lo, v8bf hi) {
    v16bf r;
#pragma unroll
    for (int i = 0; i < 8; ++i) { r[i] = lo[i]; r[i + 8] = hi[i]; }
    return r;
}

// ---------- f32 -> bf16 conversion (8 elems / thread) ----------
__global__ void f32_to_bf16_kern(const float* __restrict__ in, u16* __restrict__ out, int n) {
    const int i = blockIdx.x * blockDim.x + threadIdx.x;
    if (i * 8 >= n) return;
    const float4* f4 = (const float4*)in;
    float4 a = f4[i * 2];
    float4 b = f4[i * 2 + 1];
    union { u16 h[8]; uint4 v; } r;
    r.h[0] = f2bf(a.x); r.h[1] = f2bf(a.y); r.h[2] = f2bf(a.z); r.h[3] = f2bf(a.w);
    r.h[4] = f2bf(b.x); r.h[5] = f2bf(b.y); r.h[6] = f2bf(b.z); r.h[7] = f2bf(b.w);
    ((uint4*)out)[i] = r.v;
}

// ---------- tiled WMMA GEMM: C = act(A @ W^T + bias) ----------
// A: [M,K] bf16 row-major.  W: [N,K] bf16 row-major.  M,N % 128 == 0, K % 32 == 0.
// MODE 0: bf16 out + bias
// MODE 1: bf16 out + bias + exact GELU
// MODE 2: bf16 out + bias, store transposed (C[n][m], row stride ldo)
// MODE 3: f32 out, * scale, no bias
#define BM 128
#define BN 128
#define BK 32

template <int MODE>
__global__ __launch_bounds__(256) void gemm_bf16_wmma(
    const u16* __restrict__ A, const u16* __restrict__ W,
    const float* __restrict__ bias, void* __restrict__ Cout,
    int M, int N, int K, int ldo, float scale)
{
    __shared__ __align__(16) u16 As[2][BM * BK];   // double-buffered
    __shared__ __align__(16) u16 Ws[2][BN * BK];

    const int tid   = threadIdx.x;
    const int lane  = tid & 31;
    const int wave  = tid >> 5;        // 0..7
    const int waveM = wave >> 2;       // 0..1  -> 64 rows each
    const int waveN = wave & 3;        // 0..3  -> 32 cols each
    const int bm    = blockIdx.y * BM;
    const int bn    = blockIdx.x * BN;

    const int lrow = tid >> 1;         // 0..127 : tile row staged by this thread
    const int lcol = (tid & 1) * 16;   // 0 or 16 (bf16 elements)
    const int hi   = lane >> 4;        // 0/1 half-wave
    const int l15  = lane & 15;

    const u16* aRow = A + (size_t)(bm + lrow) * K + lcol;
    const u16* wRow = W + (size_t)(bn + lrow) * K + lcol;

    v8f acc[4][2] = {};

    auto stage = [&](int k0, int b) {
#if HAVE_ASYNC_LDS
        const u16* ag = aRow + k0;
        const u16* wg = wRow + k0;
        u16* ad = &As[b][lrow * BK + lcol];
        u16* wd = &Ws[b][lrow * BK + lcol];
        __builtin_amdgcn_global_load_async_to_lds_b128((AS1 vec4i*)ag,       (AS3 vec4i*)ad,       0, 0);
        __builtin_amdgcn_global_load_async_to_lds_b128((AS1 vec4i*)(ag + 8), (AS3 vec4i*)(ad + 8), 0, 0);
        __builtin_amdgcn_global_load_async_to_lds_b128((AS1 vec4i*)wg,       (AS3 vec4i*)wd,       0, 0);
        __builtin_amdgcn_global_load_async_to_lds_b128((AS1 vec4i*)(wg + 8), (AS3 vec4i*)(wd + 8), 0, 0);
#else
        uint4 a0 = ((const uint4*)(aRow + k0))[0];
        uint4 a1 = ((const uint4*)(aRow + k0))[1];
        uint4 w0 = ((const uint4*)(wRow + k0))[0];
        uint4 w1 = ((const uint4*)(wRow + k0))[1];
        *(uint4*)&As[b][lrow * BK + lcol]     = a0;
        *(uint4*)&As[b][lrow * BK + lcol + 8] = a1;
        *(uint4*)&Ws[b][lrow * BK + lcol]     = w0;
        *(uint4*)&Ws[b][lrow * BK + lcol + 8] = w1;
#endif
        if (k0 + BK < K) {                           // global_prefetch_b8
            __builtin_prefetch(aRow + k0 + BK, 0, 1);
            __builtin_prefetch(wRow + k0 + BK, 0, 1);
        }
    };

    const int nT = K / BK;
    stage(0, 0);

    for (int t = 0; t < nT; ++t) {
#if HAVE_ASYNC_LDS
        WAIT_ASYNC();                 // this wave's tile-t async writes landed
#endif
        __syncthreads();              // everyone's tile-t data visible; prev reads done
        if (t + 1 < nT) stage((t + 1) * BK, (t + 1) & 1);   // overlap with compute

        const u16* asb = As[t & 1];
        const u16* wsb = Ws[t & 1];

        // ---- fragments per documented CDNA5 VGPR layouts ----
        // A (16x32, MxK): lanes 0-15 row M=l15 hold K 0..7 & 16..23; lanes 16-31 hold K 8..15 & 24..31
        v16bf afrag[4];
#pragma unroll
        for (int mt = 0; mt < 4; ++mt) {
            const u16* base = &asb[(waveM * 64 + mt * 16 + l15) * BK];
            v8bf lo = *(const v8bf*)(base + hi * 8);
            v8bf h8 = *(const v8bf*)(base + 16 + hi * 8);
            afrag[mt] = cat8(lo, h8);
        }
        // B (32x16, KxN, column-major): lane n holds its column; lanes 0-15 hold K 0..7 & 16..23; lanes 16-31 hold K 8..15 & 24..31
        v16bf bfrag[2];
#pragma unroll
        for (int nt = 0; nt < 2; ++nt) {
            const u16* base = &wsb[(waveN * 32 + nt * 16 + l15) * BK];
            v8bf lo = *(const v8bf*)(base + hi * 8);
            v8bf h8 = *(const v8bf*)(base + 16 + hi * 8);
            bfrag[nt] = cat8(lo, h8);
        }
#pragma unroll
        for (int mt = 0; mt < 4; ++mt)
#pragma unroll
            for (int nt = 0; nt < 2; ++nt)
                acc[mt][nt] = __builtin_amdgcn_wmma_f32_16x16x32_bf16(
                    false, afrag[mt], false, bfrag[nt],
                    (short)0, acc[mt][nt], false, false);
    }

    // ---- epilogue ----
#pragma unroll
    for (int mt = 0; mt < 4; ++mt) {
#pragma unroll
        for (int nt = 0; nt < 2; ++nt) {
            const int col = bn + waveN * 32 + nt * 16 + l15;
#pragma unroll
            for (int v = 0; v < 8; ++v) {
                const int row = bm + waveM * 64 + mt * 16 + v + hi * 8;
                float x = acc[mt][nt][v];
                if (MODE == 0 || MODE == 1 || MODE == 2) x += bias[col];
                if (MODE == 1) x = 0.5f * x * (1.0f + erff(x * 0.7071067811865476f));
                if (MODE == 3) {
                    ((float*)Cout)[(size_t)row * ldo + col] = x * scale;
                } else if (MODE == 2) {
                    ((u16*)Cout)[(size_t)col * ldo + row] = f2bf(x);
                } else {
                    ((u16*)Cout)[(size_t)row * ldo + col] = f2bf(x);
                }
            }
        }
    }
}

// ---------- row softmax over 16384 columns (row held in registers) ----------
__global__ __launch_bounds__(256) void softmax_rows_kern(
    const float* __restrict__ S, u16* __restrict__ P)
{
    __shared__ float red[256];
    const int row = blockIdx.x;
    const int t   = threadIdx.x;
    const float* s = S + (size_t)row * 16384;

    float v[64];
    float m = -3.4e38f;
#pragma unroll
    for (int i = 0; i < 64; ++i) { v[i] = s[t + i * 256]; m = fmaxf(m, v[i]); }
    red[t] = m; __syncthreads();
    for (int off = 128; off > 0; off >>= 1) {
        if (t < off) red[t] = fmaxf(red[t], red[t + off]);
        __syncthreads();
    }
    m = red[0]; __syncthreads();

    float sum = 0.0f;
#pragma unroll
    for (int i = 0; i < 64; ++i) { v[i] = expf(v[i] - m); sum += v[i]; }
    red[t] = sum; __syncthreads();
    for (int off = 128; off > 0; off >>= 1) {
        if (t < off) red[t] += red[t + off];
        __syncthreads();
    }
    const float inv = 1.0f / red[0];

    u16* p = P + (size_t)row * 16384;
#pragma unroll
    for (int i = 0; i < 64; ++i) p[t + i * 256] = f2bf(v[i] * inv);
}

// ---------- column means of P over the 1024 (b,q) rows ----------
__global__ void col_mean_kern(const u16* __restrict__ P, float* __restrict__ cm) {
    const int k = blockIdx.x * 256 + threadIdx.x;    // 0..16383
    float a = 0.0f;
    for (int r = 0; r < 1024; ++r) a += bf2f(P[(size_t)r * 16384 + k]);
    cm[k] = a * (1.0f / 1024.0f);
}

// ---------- per-bank mean + argsort(-importance), 64 entries ----------
__global__ void finalize_kern(const float* __restrict__ cm, float* __restrict__ out_idx) {
    __shared__ float imp[64];
    const int t = threadIdx.x;                       // 64 threads
    float a = 0.0f;
    for (int i = 0; i < 256; ++i) a += cm[t * 256 + i];
    imp[t] = a * (1.0f / 256.0f);
    __syncthreads();
    if (t == 0) {
        bool used[64] = {};
        for (int i = 0; i < 64; ++i) {
            int best = 0; float bv = -3.4e38f;
            for (int j = 0; j < 64; ++j)
                if (!used[j] && imp[j] > bv) { bv = imp[j]; best = j; }
            used[best] = true;
            out_idx[i] = (float)best;
        }
    }
}

// ---------- host-side orchestration ----------
extern "C" void kernel_launch(void* const* d_in, const int* in_sizes, int n_in,
                              void* d_out, int out_size, void* d_ws, size_t ws_size,
                              hipStream_t stream)
{
    (void)in_sizes; (void)n_in; (void)out_size; (void)ws_size;
    const float* x    = (const float*)d_in[0];   // [4,256,1024]
    const float* bank = (const float*)d_in[1];   // [64,256,1024]
    const float* Wq   = (const float*)d_in[2];
    const float* bq   = (const float*)d_in[3];
    const float* Wk   = (const float*)d_in[4];
    const float* bk   = (const float*)d_in[5];
    const float* Wv   = (const float*)d_in[6];
    const float* bv   = (const float*)d_in[7];
    const float* P1   = (const float*)d_in[8];
    const float* pb1  = (const float*)d_in[9];
    const float* P2   = (const float*)d_in[10];
    const float* pb2  = (const float*)d_in[11];

    const int MQ = 1024;      // B*T_q
    const int D  = 1024;      // D_in == D_model
    const int NK = 16384;     // N * T_k

    char*  ws  = (char*)d_ws;
    size_t off = 0;
    auto carve = [&](size_t bytes) -> void* {
        void* p = ws + off;
        off += (bytes + 255) & ~(size_t)255;
        return p;
    };
    u16* Xb   = (u16*)carve((size_t)MQ * D * 2);
    u16* P1b  = (u16*)carve((size_t)D * D * 2);
    u16* P2b  = (u16*)carve((size_t)D * D * 2);
    u16* Wqb  = (u16*)carve((size_t)D * D * 2);
    u16* Wkb  = (u16*)carve((size_t)D * D * 2);
    u16* Wvb  = (u16*)carve((size_t)D * D * 2);
    u16* Hb   = (u16*)carve((size_t)MQ * D * 2);
    u16* Qib  = (u16*)carve((size_t)MQ * D * 2);
    u16* Qb   = (u16*)carve((size_t)MQ * D * 2);
    u16* Kb   = (u16*)carve((size_t)NK * D * 2);   // later reused as P (bf16)
    u16* Vtb  = (u16*)carve((size_t)D * NK * 2);   // V transposed: [D][NK]
    float* cm = (float*)carve((size_t)NK * 4);
    float* Sf = (float*)carve((size_t)MQ * NK * 4); // scores f32; head aliases dead bank-bf16
    u16* bankb = (u16*)Sf;                          // bank bf16 (dead before S is written)
    u16* Pb    = Kb;                                // P bf16 overlays dead K

    const dim3 blk(256);
    f32_to_bf16_kern<<<(MQ * D) / 2048, blk, 0, stream>>>(x,    Xb,  MQ * D);
    f32_to_bf16_kern<<<(D * D)  / 2048, blk, 0, stream>>>(P1,   P1b, D * D);
    f32_to_bf16_kern<<<(D * D)  / 2048, blk, 0, stream>>>(P2,   P2b, D * D);
    f32_to_bf16_kern<<<(D * D)  / 2048, blk, 0, stream>>>(Wq,   Wqb, D * D);
    f32_to_bf16_kern<<<(D * D)  / 2048, blk, 0, stream>>>(Wk,   Wkb, D * D);
    f32_to_bf16_kern<<<(D * D)  / 2048, blk, 0, stream>>>(Wv,   Wvb, D * D);
    f32_to_bf16_kern<<<(NK * D) / 2048, blk, 0, stream>>>(bank, bankb, NK * D);

    const dim3 gSq(D / BN, MQ / BM);       // 8 x 8
    const dim3 gKV(D / BN, NK / BM);       // 8 x 128
    const dim3 gSc(NK / BN, MQ / BM);      // 128 x 8

    // MLP projector + Q projection
    gemm_bf16_wmma<1><<<gSq, blk, 0, stream>>>(Xb,  P1b, pb1, Hb,  MQ, D, D, D, 1.0f);
    gemm_bf16_wmma<0><<<gSq, blk, 0, stream>>>(Hb,  P2b, pb2, Qib, MQ, D, D, D, 1.0f);
    gemm_bf16_wmma<0><<<gSq, blk, 0, stream>>>(Qib, Wqb, bq,  Qb,  MQ, D, D, D, 1.0f);
    // K / V projections (V stored transposed for the context GEMM)
    gemm_bf16_wmma<0><<<gKV, blk, 0, stream>>>(bankb, Wkb, bk, Kb,  NK, D, D, D,  1.0f);
    gemm_bf16_wmma<2><<<gKV, blk, 0, stream>>>(bankb, Wvb, bv, Vtb, NK, D, D, NK, 1.0f);
    // scores = Q K^T * 1/sqrt(1024)  (f32, lives in the 192MB L2)
    gemm_bf16_wmma<3><<<gSc, blk, 0, stream>>>(Qb, Kb, nullptr, Sf, MQ, NK, D, NK, 0.03125f);
    // softmax rows -> P (bf16)
    softmax_rows_kern<<<MQ, blk, 0, stream>>>(Sf, Pb);
    // context = P @ V  ==  P @ (Vt)^T
    gemm_bf16_wmma<3><<<gSq, blk, 0, stream>>>(Pb, Vtb, nullptr, (float*)d_out,
                                               MQ, D, NK, D, 1.0f);
    // importance: column means -> per-bank means -> argsort
    col_mean_kern<<<NK / 256, blk, 0, stream>>>(Pb, cm);
    finalize_kern<<<1, 64, 0, stream>>>(cm, (float*)d_out + (size_t)MQ * D);
}
